// MCABlock_68856915689525
// MI455X (gfx1250) — matrix operand order repaired and checked by reading, not compiled, not run on hardware
//
#include <hip/hip_runtime.h>
#include <hip/hip_bf16.h>

typedef __attribute__((ext_vector_type(16))) _Float16 v16h;
typedef __attribute__((ext_vector_type(8)))  _Float16 v8h;
typedef __attribute__((ext_vector_type(8)))  float    v8f;

#define BB 32
#define NN 2000
#define BN 64000

static constexpr float IN_EPS_C = 1e-3f;
static constexpr float BN_EPS_C = 1e-5f;

// ------------------------------------------------------------------
// conv1: CIN=4 -> 128, data (B,4,N,1) -> point-major out[p][128]
// ------------------------------------------------------------------
__global__ void k_conv_in(const float* __restrict__ data, const float* __restrict__ w,
                          const float* __restrict__ bias, float* __restrict__ out) {
  int idx = blockIdx.x * blockDim.x + threadIdx.x;     // p*128 + o
  if (idx >= BN * 128) return;
  int o = idx & 127;
  int p = idx >> 7;
  int b = p / NN, n = p - b * NN;
  const float* dp = data + (size_t)(b * 4) * NN + n;
  float acc = bias[o];
#pragma unroll
  for (int c = 0; c < 4; ++c) acc += w[o * 4 + c] * dp[(size_t)c * NN];
  out[idx] = acc;
}

// ------------------------------------------------------------------
// one-time fp32 -> f16 weight convert (loop-invariant across tiles)
// ------------------------------------------------------------------
__global__ void k_wcvt(const float* __restrict__ w, _Float16* __restrict__ wh, int n) {
  int i = blockIdx.x * blockDim.x + threadIdx.x;
  if (i < n) wh[i] = (_Float16)w[i];
}

// ------------------------------------------------------------------
// per-(b,c) instance-norm statistics over N points (point-major input)
// ------------------------------------------------------------------
__global__ void k_stats(const float* __restrict__ x, int C,
                        float* __restrict__ mean, float* __restrict__ var) {
  __shared__ float ssum[1024];
  __shared__ float ssq[1024];
  int b = blockIdx.x;
  int tid = threadIdx.x;
  int tpc = 1024 / C;                 // threads per channel
  int c = tid % C, slot = tid / C;
  float s = 0.f, q = 0.f;
  const float* xp = x + (size_t)b * NN * C + c;
  for (int n = slot; n < NN; n += tpc) {
    float v = xp[(size_t)n * C];
    s += v; q += v * v;
  }
  ssum[tid] = s; ssq[tid] = q;
  __syncthreads();
  if (tid < C) {
    float S = 0.f, Q = 0.f;
    for (int k = 0; k < tpc; ++k) { S += ssum[tid + k * C]; Q += ssq[tid + k * C]; }
    float m = S * (1.0f / NN);
    float v = Q * (1.0f / NN) - m * m;
    mean[b * C + tid] = m;
    var[b * C + tid]  = v < 0.f ? 0.f : v;
  }
}

// ------------------------------------------------------------------
// analytic bnorm-of-inorm: bn_mean==0, bn_var = mean_b[ v/(v+IN_EPS) ].
// Produces per-(b,c) affine: out = relu(a*x + d)
// ------------------------------------------------------------------
__global__ void k_coeff(const float* __restrict__ mean, const float* __restrict__ var,
                        const float* __restrict__ g, const float* __restrict__ be,
                        int C, float* __restrict__ ca, float* __restrict__ cd) {
  __shared__ float sw[BB];
  __shared__ float s_bnvar;
  int c = blockIdx.x, b = threadIdx.x;
  float v = var[b * C + c];
  sw[b] = v / (v + IN_EPS_C);
  __syncthreads();
  if (b == 0) {
    float S = 0.f;
    for (int i = 0; i < BB; ++i) S += sw[i];
    s_bnvar = S * (1.0f / BB);
  }
  __syncthreads();
  float inv_in = rsqrtf(v + IN_EPS_C);
  float scale  = g[c] * rsqrtf(s_bnvar + BN_EPS_C);
  float a = inv_in * scale;
  ca[b * C + c] = a;
  cd[b * C + c] = be[c] - mean[b * C + c] * a;
}

// ------------------------------------------------------------------
// apply affine + relu + fp32->f16 (WMMA feed), point-major
// ------------------------------------------------------------------
__global__ void k_apply(const float* __restrict__ x, const float* __restrict__ ca,
                        const float* __restrict__ cd, int C, _Float16* __restrict__ xh) {
  int idx = blockIdx.x * blockDim.x + threadIdx.x;
  if (idx >= BN * C) return;
  int c = idx % C;
  int p = idx / C;
  int b = p / NN;
  float v = ca[b * C + c] * x[idx] + cd[b * C + c];
  xh[idx] = (_Float16)(v > 0.f ? v : 0.f);
}

// ------------------------------------------------------------------
// WMMA GEMM: out[p][O] = alpha*(Wh[O][K] * xh[p][K] + bias) + add[p][O]
// MT M-tiles per wave sharing one B fragment; 8 waves/block.
// MT=2: grid (BN/64, O/64);  MT=1: grid (BN/64, O/32)
// ------------------------------------------------------------------
template <int MT>
__global__ void k_gemm_wmma(const _Float16* __restrict__ Xh, const _Float16* __restrict__ Wh,
                            const float* __restrict__ bias, const float* __restrict__ add,
                            const float* __restrict__ alpha_p, float* __restrict__ out,
                            int K, int O) {
  int lane = threadIdx.x & 31;
  int wv   = threadIdx.x >> 5;
  int hi   = lane >> 4;
  int l16  = lane & 15;
  int tile_n = blockIdx.x * 4 + (wv & 3);
  int tm     = blockIdx.y * 2 + (wv >> 2);
  int nG     = tile_n * 16 + l16;         // B/D column (point)
  int mBase  = tm * 16 * MT;

  if (add) __builtin_prefetch((const void*)(add + (size_t)nG * O + mBase), 0, 3);

  v8f acc[MT] = {};
  for (int k0 = 0; k0 < K; k0 += 32) {
    // B fragment (32x16 f16): lane half selects k-block, contiguous 32B load
    const _Float16* xp = Xh + (size_t)nG * K + k0 + (hi ? 16 : 0);
    v16h bfr = *(const v16h*)xp;
#pragma unroll
    for (int t = 0; t < MT; ++t) {
      // A fragment (16x32 f16): elems 0..15 -> k0+off0..k0+off0+15 (contiguous, matches B)
      const _Float16* wp = Wh + (size_t)(mBase + t * 16 + l16) * K + k0 + (hi ? 16 : 0);
      v8h alo = *(const v8h*)wp;
      v8h ahi = *(const v8h*)(wp + 8);
      v16h a;
#pragma unroll
      for (int e = 0; e < 8; ++e) { a[e] = alo[e]; a[8 + e] = ahi[e]; }
      acc[t] = __builtin_amdgcn_wmma_f32_16x16x32_f16(false, a, false, bfr,
                                                      (short)0, acc[t], false, false);
    }
  }
  float alpha = alpha_p ? alpha_p[0] : 1.0f;
#pragma unroll
  for (int t = 0; t < MT; ++t) {
    int m0 = mBase + t * 16 + (hi ? 8 : 0);
    size_t o0 = (size_t)nG * O + m0;
#pragma unroll
    for (int r = 0; r < 8; ++r) {
      float v = alpha * (acc[t][r] + bias[m0 + r]);
      if (add) v += add[o0 + r];
      out[o0 + r] = v;
    }
  }
}

// ------------------------------------------------------------------
// final O=1 conv + relu(tanh): one wave per point, shuffle reduce
// ------------------------------------------------------------------
__global__ void k_logits(const float* __restrict__ x, const float* __restrict__ w,
                         const float* __restrict__ b0, float* __restrict__ logits,
                         float* __restrict__ w8) {
  int lane = threadIdx.x & 31;
  int wv   = threadIdx.x >> 5;
  int p = blockIdx.x * 8 + wv;
  if (p >= BN) return;
  float4 xv = ((const float4*)(x + (size_t)p * 128))[lane];
  float4 wv4 = ((const float4*)w)[lane];
  float s = xv.x * wv4.x + xv.y * wv4.y + xv.z * wv4.z + xv.w * wv4.w;
#pragma unroll
  for (int off = 16; off > 0; off >>= 1) s += __shfl_down(s, off, 32);
  if (lane == 0) {
    float lg = s + b0[0];
    logits[p] = lg;
    float t = tanhf(lg);
    w8[p] = t > 0.f ? t : 0.f;
  }
}

// ------------------------------------------------------------------
// XwX[b] = sum_n w8 * X X^T  (9x9 symmetric, 45 unique, LDS atomics)
// ------------------------------------------------------------------
__global__ void k_xwx(const float* __restrict__ xs, const float* __restrict__ w8,
                      float* __restrict__ xwx) {
  __shared__ float sh[45];
  int b = blockIdx.x, tid = threadIdx.x;
  if (tid < 45) sh[tid] = 0.f;
  __syncthreads();
  float acc[45];
#pragma unroll
  for (int k = 0; k < 45; ++k) acc[k] = 0.f;
  for (int n = tid; n < NN; n += blockDim.x) {
    float4 xv = *(const float4*)(xs + ((size_t)b * NN + n) * 4);
    float w = w8[b * NN + n];
    float X[9] = {xv.z * xv.x, xv.z * xv.y, xv.z,
                  xv.w * xv.x, xv.w * xv.y, xv.w,
                  xv.x,        xv.y,        1.f};
    int k = 0;
#pragma unroll
    for (int i = 0; i < 9; ++i)
#pragma unroll
      for (int j = i; j < 9; ++j)
        acc[k++] += w * X[i] * X[j];
  }
#pragma unroll
  for (int k = 0; k < 45; ++k) atomicAdd(&sh[k], acc[k]);
  __syncthreads();
  if (tid < 81) {
    int i = tid / 9, j = tid % 9;
    int ii = i < j ? i : j, jj = i < j ? j : i;
    int k = ii * (19 - ii) / 2 + (jj - ii);
    xwx[b * 81 + tid] = sh[k];
  }
}

// ------------------------------------------------------------------
// 9x9 symmetric Jacobi eigensolve, one lane per batch element
// ------------------------------------------------------------------
__global__ void k_eigh9(const float* __restrict__ xwx, float* __restrict__ ehat_out,
                        float* __restrict__ ehat_ws) {
  int b = threadIdx.x;
  if (b >= BB) return;
  float A[81], V[81];
  for (int i = 0; i < 81; ++i) { A[i] = xwx[b * 81 + i]; V[i] = 0.f; }
  for (int i = 0; i < 9; ++i) V[i * 9 + i] = 1.f;
  for (int sweep = 0; sweep < 24; ++sweep) {
    for (int p = 0; p < 8; ++p) {
      for (int q = p + 1; q < 9; ++q) {
        float apq = A[p * 9 + q];
        if (fabsf(apq) < 1e-28f) continue;
        float app = A[p * 9 + p], aqq = A[q * 9 + q];
        float tau = (aqq - app) / (2.f * apq);
        float t = (tau >= 0.f ? 1.f : -1.f) / (fabsf(tau) + sqrtf(1.f + tau * tau));
        float c = rsqrtf(1.f + t * t);
        float s = t * c;
        for (int k = 0; k < 9; ++k) {
          float akp = A[k * 9 + p], akq = A[k * 9 + q];
          A[k * 9 + p] = c * akp - s * akq;
          A[k * 9 + q] = s * akp + c * akq;
        }
        for (int k = 0; k < 9; ++k) {
          float apk = A[p * 9 + k], aqk = A[q * 9 + k];
          A[p * 9 + k] = c * apk - s * aqk;
          A[q * 9 + k] = s * apk + c * aqk;
        }
        for (int k = 0; k < 9; ++k) {
          float vkp = V[k * 9 + p], vkq = V[k * 9 + q];
          V[k * 9 + p] = c * vkp - s * vkq;
          V[k * 9 + q] = s * vkp + c * vkq;
        }
      }
    }
  }
  int jm = 0; float dm = A[0];
  for (int j = 1; j < 9; ++j) { float d = A[j * 9 + j]; if (d < dm) { dm = d; jm = j; } }
  float nrm = 0.f;
  for (int i = 0; i < 9; ++i) { float e = V[i * 9 + jm]; nrm += e * e; }
  float inv = rsqrtf(nrm + 1e-30f);
  for (int i = 0; i < 9; ++i) {
    float e = V[i * 9 + jm] * inv;
    ehat_out[b * 9 + i] = e;
    ehat_ws[b * 9 + i]  = e;
  }
}

// ------------------------------------------------------------------
// epipolar residuals
// ------------------------------------------------------------------
__global__ void k_residual(const float* __restrict__ xs, const float* __restrict__ ehat,
                           float* __restrict__ res) {
  int p = blockIdx.x * blockDim.x + threadIdx.x;
  if (p >= BN) return;
  int b = p / NN;
  const float* F = ehat + b * 9;
  float4 xv = *(const float4*)(xs + (size_t)p * 4);
  float Fx0 = F[0] * xv.x + F[1] * xv.y + F[2];
  float Fx1 = F[3] * xv.x + F[4] * xv.y + F[5];
  float Fx2 = F[6] * xv.x + F[7] * xv.y + F[8];
  float Ft0 = F[0] * xv.z + F[3] * xv.w + F[6];
  float Ft1 = F[1] * xv.z + F[4] * xv.w + F[7];
  float x2Fx1 = xv.z * Fx0 + xv.w * Fx1 + Fx2;
  res[p] = x2Fx1 * x2Fx1 * (1.f / (Fx0 * Fx0 + Fx1 * Fx1 + 1e-15f) +
                            1.f / (Ft0 * Ft0 + Ft1 * Ft1 + 1e-15f));
}

// ==================================================================
// host orchestration
// ==================================================================
struct WsPlan {
  float *A, *B, *H128, *P, *Q, *R, *H64, *S3, *T3, *H32;
  _Float16* XH;
  _Float16* WH;
  float *STATM, *STATV, *CA, *CD, *W8, *XWX, *EH;
};

static void norm_apply(hipStream_t st, const WsPlan& w, const float* x, int C,
                       const float* g, const float* be) {
  k_stats<<<dim3(BB), dim3(1024), 0, st>>>(x, C, w.STATM, w.STATV);
  k_coeff<<<dim3(C), dim3(BB), 0, st>>>(w.STATM, w.STATV, g, be, C, w.CA, w.CD);
  int tot = BN * C;
  k_apply<<<dim3((tot + 255) / 256), dim3(256), 0, st>>>(x, w.CA, w.CD, C, w.XH);
}

static void gemm(hipStream_t st, const WsPlan& w, int K, const float* W, const float* bias,
                 int O, const float* add, const float* alpha, float* out) {
  int wn = O * K;
  k_wcvt<<<dim3((wn + 255) / 256), dim3(256), 0, st>>>(W, w.WH, wn);
  if (O >= 64) {
    dim3 grid(BN / 64, O / 64);
    k_gemm_wmma<2><<<grid, dim3(256), 0, st>>>(w.XH, w.WH, bias, add, alpha, out, K, O);
  } else {
    dim3 grid(BN / 64, O / 32);
    k_gemm_wmma<1><<<grid, dim3(256), 0, st>>>(w.XH, w.WH, bias, add, alpha, out, K, O);
  }
}

static void nrc_op(hipStream_t st, const WsPlan& w, const float* x, int Cin,
                   const float* g, const float* be, const float* Wt, const float* bias,
                   int Cout, const float* add, const float* alpha, float* out) {
  norm_apply(st, w, x, Cin, g, be);
  gemm(st, w, Cin, Wt, bias, Cout, add, alpha, out);
}

static void pointcn_op(hipStream_t st, const WsPlan& w, const float* x, int C,
                       void* const* d_in, int pi, float* h, float* out) {
  auto Fp = [&](int i) { return (const float*)d_in[i]; };
  nrc_op(st, w, x, C, Fp(pi + 0), Fp(pi + 1), Fp(pi + 2), Fp(pi + 3), C, nullptr, nullptr, h);
  nrc_op(st, w, h, C, Fp(pi + 4), Fp(pi + 5), Fp(pi + 6), Fp(pi + 7), C, x, nullptr, out);
}

extern "C" void kernel_launch(void* const* d_in, const int* in_sizes, int n_in,
                              void* d_out, int out_size, void* d_ws, size_t ws_size,
                              hipStream_t stream) {
  (void)in_sizes; (void)n_in; (void)out_size; (void)ws_size;
  auto Fp = [&](int i) { return (const float*)d_in[i]; };
  const float* data = Fp(0);
  const float* xs   = Fp(1);
  float* out = (float*)d_out;           // [logits 64000][ehat 288][residual 64000]

  const size_t S128 = (size_t)BN * 128, S64 = (size_t)BN * 64, S32 = (size_t)BN * 32;
  float* ws = (float*)d_ws;
  size_t off = 0;
  auto alloc = [&](size_t n) { float* p = ws + off; off += n; return p; };
  WsPlan w;
  w.A = alloc(S128); w.B = alloc(S128); w.H128 = alloc(S128);
  w.P = alloc(S64);  w.Q = alloc(S64);  w.R = alloc(S64); w.H64 = alloc(S64);
  w.S3 = alloc(S32); w.T3 = alloc(S32); w.H32 = alloc(S32);
  w.XH = (_Float16*)alloc(S128 / 2);
  w.WH = (_Float16*)alloc(128 * 128 / 2);
  w.STATM = alloc(BB * 128); w.STATV = alloc(BB * 128);
  w.CA = alloc(BB * 128);    w.CD = alloc(BB * 128);
  w.W8 = alloc(BN); w.XWX = alloc(BB * 81); w.EH = alloc(BB * 9);

  // x_row -> A
  k_conv_in<<<dim3((BN * 128 + 255) / 256), dim3(256), 0, stream>>>(data, Fp(2), Fp(3), w.A);

  pointcn_op(stream, w, w.A, 128, d_in, 4, w.H128, w.B);                     // x1_1 -> B
  nrc_op(stream, w, w.B, 128, Fp(100), Fp(101), Fp(102), Fp(103), 64,
         nullptr, nullptr, w.P);                                             // x2_0 -> P
  pointcn_op(stream, w, w.B, 128, d_in, 12, w.H128, w.A);                    // x1_2 -> A
  pointcn_op(stream, w, w.P, 64, d_in, 52, w.H64, w.Q);                      // x2_1 -> Q
  // t13in = x1_2 + s*conv(norm(x2_1), info2_1) -> B
  nrc_op(stream, w, w.Q, 64, Fp(113), Fp(114), Fp(115), Fp(116), 128, w.A, Fp(117), w.B);
  // t22in = x2_1 + s*conv(norm(x1_2), info1_2) -> R
  nrc_op(stream, w, w.A, 128, Fp(108), Fp(109), Fp(110), Fp(111), 64, w.Q, Fp(112), w.R);
  pointcn_op(stream, w, w.B, 128, d_in, 20, w.H128, w.A);                    // x1_3 -> A
  pointcn_op(stream, w, w.R, 64, d_in, 60, w.H64, w.Q);                      // x2_2 -> Q
  pointcn_op(stream, w, w.A, 128, d_in, 28, w.H128, w.B);                    // x1_4 -> B
  pointcn_op(stream, w, w.Q, 64, d_in, 68, w.H64, w.P);                      // x2_3 -> P
  // t15in = x1_4 + s*conv(norm(x2_3), iinfo2_1) -> A
  nrc_op(stream, w, w.P, 64, Fp(123), Fp(124), Fp(125), Fp(126), 128, w.B, Fp(127), w.A);
  // t24in = x2_3 + s*conv(norm(x1_4), iinfo1_2) -> R
  nrc_op(stream, w, w.B, 128, Fp(118), Fp(119), Fp(120), Fp(121), 64, w.P, Fp(122), w.R);
  pointcn_op(stream, w, w.A, 128, d_in, 36, w.H128, w.B);                    // x1_5 -> B
  pointcn_op(stream, w, w.R, 64, d_in, 76, w.H64, w.Q);                      // x2_4 -> Q
  nrc_op(stream, w, w.Q, 64, Fp(104), Fp(105), Fp(106), Fp(107), 32,
         nullptr, nullptr, w.S3);                                            // x3_0 -> S3
  pointcn_op(stream, w, w.B, 128, d_in, 44, w.H128, w.A);                    // x1_6 -> A
  pointcn_op(stream, w, w.Q, 64, d_in, 84, w.H64, w.P);                      // x2_5 -> P
  pointcn_op(stream, w, w.S3, 32, d_in, 92, w.H32, w.T3);                    // x3_1 -> T3
  // out = x1_6 + s*conv(norm(x2_5), iiinfo2_1) -> B
  nrc_op(stream, w, w.P, 64, Fp(128), Fp(129), Fp(130), Fp(131), 128, w.A, Fp(132), w.B);
  // out += s*conv(norm(x3_1), iiinfo3_1) -> A  (final)
  nrc_op(stream, w, w.T3, 32, Fp(133), Fp(134), Fp(135), Fp(136), 128, w.B, Fp(137), w.A);

  // logits + weights
  k_logits<<<dim3(BN / 8), dim3(256), 0, stream>>>(w.A, Fp(138), Fp(139), out, w.W8);
  // weighted 8-point: XwX, eigh, residuals
  k_xwx<<<dim3(BB), dim3(256), 0, stream>>>(xs, w.W8, w.XWX);
  k_eigh9<<<dim3(1), dim3(BB), 0, stream>>>(w.XWX, out + BN, w.EH);
  k_residual<<<dim3((BN + 255) / 256), dim3(256), 0, stream>>>(xs, w.EH, out + BN + BB * 9);
}
